// GNN_87677462380643
// MI455X (gfx1250) — matrix operand order repaired and checked by reading, not compiled, not run on hardware
//
#include <hip/hip_runtime.h>
#include <hip/hip_bf16.h>

// ---------------------------------------------------------------------------
// GraphSAGE (2x SAGEConv + global_mean_pool) for MI455X / gfx1250.
// Memory-bound workload: scatter atomics dominate; GEMMs done with
// v_wmma_f32_16x16x32_bf16. Each wave owns a 16-row x 128-col output stripe:
// A fragments built once per K-slab (single v_perm_b32 bf16 packing) and
// reused across 8 column tiles -> 16 WMMAs per A conversion.
// ---------------------------------------------------------------------------

typedef __attribute__((ext_vector_type(16))) __bf16       v16bf;
typedef __attribute__((ext_vector_type(8)))  float        v8f;
typedef __attribute__((ext_vector_type(8)))  unsigned int v8u;

#define GNN_D 128   // feature width (D == H == O)
#define GNN_G 64    // number of graphs for pooling

// ---- fp32 -> bf16 pack, truncating: one v_perm_b32 ------------------------
__device__ __forceinline__ unsigned int packbf2_trunc(float lo, float hi) {
  return __builtin_amdgcn_perm(__builtin_bit_cast(unsigned int, hi),
                               __builtin_bit_cast(unsigned int, lo),
                               0x07060302u);
}

// ---- fp32 -> bf16 round-to-nearest-even (one-time weight repack only) -----
__device__ __forceinline__ unsigned int f2bf1(float f) {
  unsigned int u = __builtin_bit_cast(unsigned int, f);
  return (u + 0x7FFFu + ((u >> 16) & 1u)) >> 16;
}
__device__ __forceinline__ unsigned int packbf2_rne(float lo, float hi) {
  return (f2bf1(hi) << 16) | (f2bf1(lo) & 0xFFFFu);
}

__device__ __forceinline__ void atomic_add_f32(float* p, float v) {
  // relaxed + agent scope -> native global_atomic_add_f32 on gfx1250
  __hip_atomic_fetch_add(p, v, __ATOMIC_RELAXED, __HIP_MEMORY_SCOPE_AGENT);
}

// Build one 16x32 bf16 A fragment (ISA 7.12.2 layout) from a row-major fp32
// row pointer, optionally scaled. lane row = lane&15; K chunks at
// kt*32 + khalf*8 and kt*32 + 16 + khalf*8.
__device__ __forceinline__ v16bf make_afrag(const float* __restrict__ p,
                                            int kb, float s) {
  float4 f0 = *(const float4*)(p + kb);
  float4 f1 = *(const float4*)(p + kb + 4);
  float4 f2 = *(const float4*)(p + kb + 16);
  float4 f3 = *(const float4*)(p + kb + 20);
  v8u u;
  u[0] = packbf2_trunc(f0.x * s, f0.y * s);
  u[1] = packbf2_trunc(f0.z * s, f0.w * s);
  u[2] = packbf2_trunc(f1.x * s, f1.y * s);
  u[3] = packbf2_trunc(f1.z * s, f1.w * s);
  u[4] = packbf2_trunc(f2.x * s, f2.y * s);
  u[5] = packbf2_trunc(f2.z * s, f2.w * s);
  u[6] = packbf2_trunc(f3.x * s, f3.y * s);
  u[7] = packbf2_trunc(f3.z * s, f3.w * s);
  return __builtin_bit_cast(v16bf, u);
}

// ---------------------------------------------------------------------------
// Repack a 128x128 row-major fp32 weight into WMMA B-fragment layout (bf16).
// Layout: [kt 0..3][nt 0..7][lane 0..31][8 x u32 (16 bf16)]
//   lane: col = lane&15, kbase = (lane>>4)*8
//   u32 j2 holds K = kt*32 + kbase + 2*(j2&3) + (j2>>2)*16 (lo) and +1 (hi), column = col.
// ---------------------------------------------------------------------------
__global__ void gnn_wb_pack(const float* __restrict__ W,
                            unsigned int* __restrict__ out) {
  int q = blockIdx.x * blockDim.x + threadIdx.x;  // 4*8*32*8 = 8192 u32
  if (q >= 4 * 8 * 32 * 8) return;
  int j2   = q & 7;
  int lane = (q >> 3) & 31;
  int nt   = (q >> 8) & 7;
  int kt   = q >> 11;
  int c     = lane & 15;
  int kbase = (lane >> 4) * 8;
  int k0    = kt * 32 + kbase + 2 * (j2 & 3) + (j2 >> 2) * 16;
  int n     = nt * 16 + c;
  out[q] = packbf2_rne(W[k0 * GNN_D + n], W[(k0 + 1) * GNN_D + n]);
}

// ---- degree / graph-count / reciprocal ------------------------------------
__global__ void gnn_degree(const int* __restrict__ dst, int* __restrict__ cnt,
                           int E) {
  int e = blockIdx.x * blockDim.x + threadIdx.x;
  if (e < E) atomicAdd(&cnt[dst[e]], 1);
}
__global__ void gnn_gcnt(const int* __restrict__ batch, int* __restrict__ gcnt,
                         int N) {
  int i = blockIdx.x * blockDim.x + threadIdx.x;
  if (i < N) atomicAdd(&gcnt[batch[i]], 1);
}
__global__ void gnn_inv(const int* __restrict__ cnt, float* __restrict__ inv,
                        int N) {
  int i = blockIdx.x * blockDim.x + threadIdx.x;
  if (i < N) inv[i] = 1.0f / fmaxf((float)cnt[i], 1.0f);
}

// ---------------------------------------------------------------------------
// Edge scatter: agg[dst] += feat[src].  One wave per edge: lane handles one
// float4 chunk of the 128-wide feature row (coalesced gather, L2 atomics).
// ---------------------------------------------------------------------------
__global__ void gnn_scatter(const float* __restrict__ feat,
                            const int* __restrict__ src,
                            const int* __restrict__ dst,
                            float* __restrict__ agg, int E) {
  long long u = (long long)blockIdx.x * blockDim.x + threadIdx.x;
  if (u >= (long long)E * 32) return;
  int e = (int)(u >> 5);
  int c = ((int)u & 31) << 2;
  const float4 v = *(const float4*)(feat + (size_t)src[e] * GNN_D + c);
  float* p = agg + (size_t)dst[e] * GNN_D + c;
  atomic_add_f32(p + 0, v.x);
  atomic_add_f32(p + 1, v.y);
  atomic_add_f32(p + 2, v.z);
  atomic_add_f32(p + 3, v.w);
}

// ---------------------------------------------------------------------------
// Fused SAGE layer GEMM:  out = act( (agg .* inv) @ Wl + b + self @ Wr )
// One wave = one 16x128 output stripe (8 column tiles, v8f acc each).
// blockDim=256 (8 waves); wave w -> row tile tm = blockIdx.x*8 + w.
// K=128 in 4 slabs of 32: per slab build 2 A fragments (agg path scaled by
// 1/deg, self path) then issue 16 WMMAs reusing them across column tiles.
// If pool!=0: epilogue atomically accumulates into gsum[batch[row]*128+n].
// ---------------------------------------------------------------------------
__global__ __launch_bounds__(256)
void gnn_sage_gemm(const float* __restrict__ Aagg, const float* __restrict__ inv,
                   const float* __restrict__ Aself,
                   const unsigned int* __restrict__ WBl,
                   const unsigned int* __restrict__ WBr,
                   const float* __restrict__ bias,
                   float* __restrict__ out,
                   const int* __restrict__ batch, float* __restrict__ gsum,
                   int N, int relu, int pool) {
  const int lane  = threadIdx.x & 31;
  const int wv    = threadIdx.x >> 5;       // wave in block
  const int tm    = blockIdx.x * 8 + wv;    // row tile
  const int r     = lane & 15;
  const int khalf = lane >> 4;

  int row  = tm * 16 + r;
  int rowc = row < N ? row : N - 1;         // branchless clamp: EXEC all-ones
  const float s = inv[rowc];
  const float* pa = Aagg  + (size_t)rowc * GNN_D;
  const float* px = Aself + (size_t)rowc * GNN_D;

  v8f acc[8] = {};
#pragma unroll
  for (int kt = 0; kt < 4; ++kt) {
    const int kb = kt * 32 + khalf * 8;
    v16bf afragA = make_afrag(pa, kb, s);      // mean-aggregated, scaled
    v16bf afragX = make_afrag(px, kb, 1.0f);   // self features

    const unsigned int* bl = WBl + (size_t)kt * (8 * 32 * 8) + (size_t)lane * 8;
    const unsigned int* br = WBr + (size_t)kt * (8 * 32 * 8) + (size_t)lane * 8;
#pragma unroll
    for (int nt = 0; nt < 8; ++nt) {
      v8u ubl = *(const v8u*)(bl + (size_t)nt * (32 * 8));
      acc[nt] = __builtin_amdgcn_wmma_f32_16x16x32_bf16(
          false, afragA, false, __builtin_bit_cast(v16bf, ubl), (short)0,
          acc[nt], false, false);
      v8u ubr = *(const v8u*)(br + (size_t)nt * (32 * 8));
      acc[nt] = __builtin_amdgcn_wmma_f32_16x16x32_bf16(
          false, afragX, false, __builtin_bit_cast(v16bf, ubr), (short)0,
          acc[nt], false, false);
    }
  }

  // ---- epilogue: C/D layout (32-bit 16x16): n = lane&15, m = khalf*8 + v ---
#pragma unroll
  for (int nt = 0; nt < 8; ++nt) {
    const int   n = nt * 16 + (lane & 15);
    const float b = bias[n];
#pragma unroll
    for (int v = 0; v < 8; ++v) {
      int m    = khalf * 8 + v;
      int grow = tm * 16 + m;
      float val = acc[nt][v] + b;
      if (relu) val = fmaxf(val, 0.0f);
      if (grow < N) {
        if (pool) {
          atomic_add_f32(&gsum[(size_t)batch[grow] * GNN_D + n], val);
        } else {
          out[(size_t)grow * GNN_D + n] = val;
        }
      }
    }
  }
}

__global__ void gnn_finalize(const float* __restrict__ gsum,
                             const int* __restrict__ gcnt,
                             float* __restrict__ out) {
  int t = blockIdx.x * blockDim.x + threadIdx.x;
  if (t < GNN_G * GNN_D)
    out[t] = gsum[t] / fmaxf((float)gcnt[t >> 7], 1.0f);
}

// ---------------------------------------------------------------------------
extern "C" void kernel_launch(void* const* d_in, const int* in_sizes, int n_in,
                              void* d_out, int out_size, void* d_ws,
                              size_t ws_size, hipStream_t stream) {
  (void)n_in; (void)out_size; (void)ws_size;
  const float* x    = (const float*)d_in[0];
  const int*   eix  = (const int*)d_in[1];
  const int*   bat  = (const int*)d_in[2];
  const float* W1l  = (const float*)d_in[3];
  const float* b1   = (const float*)d_in[4];
  const float* W1r  = (const float*)d_in[5];
  const float* W2l  = (const float*)d_in[6];
  const float* b2   = (const float*)d_in[7];
  const float* W2r  = (const float*)d_in[8];

  const int N = in_sizes[0] / GNN_D;
  const int E = in_sizes[1] / 2;
  const int* src = eix;
  const int* dst = eix + E;

  // ---- workspace carve-up (256B aligned) ----
  char* ws = (char*)d_ws;
  size_t off = 0;
  auto carve = [&](size_t bytes) -> void* {
    void* p = ws + off;
    off = (off + bytes + 255) & ~(size_t)255;
    return p;
  };
  float*        agg  = (float*)carve((size_t)N * GNN_D * 4);
  float*        h1   = (float*)carve((size_t)N * GNN_D * 4);
  int*          cnt  = (int*)carve((size_t)N * 4);
  float*        inv  = (float*)carve((size_t)N * 4);
  float*        gsum = (float*)carve(GNN_G * GNN_D * 4);
  int*          gcnt = (int*)carve(GNN_G * 4);
  unsigned int* wb1l = (unsigned int*)carve(8192 * 4);
  unsigned int* wb1r = (unsigned int*)carve(8192 * 4);
  unsigned int* wb2l = (unsigned int*)carve(8192 * 4);
  unsigned int* wb2r = (unsigned int*)carve(8192 * 4);

  // ---- zero accumulators (graph-capture-safe async memsets) ----
  hipMemsetAsync(agg, 0, (size_t)N * GNN_D * 4, stream);
  hipMemsetAsync(cnt, 0, (size_t)N * 4, stream);
  hipMemsetAsync(gsum, 0, GNN_G * GNN_D * 4, stream);
  hipMemsetAsync(gcnt, 0, GNN_G * 4, stream);

  // ---- repack weights to WMMA B-fragment layout ----
  gnn_wb_pack<<<32, 256, 0, stream>>>(W1l, wb1l);
  gnn_wb_pack<<<32, 256, 0, stream>>>(W1r, wb1r);
  gnn_wb_pack<<<32, 256, 0, stream>>>(W2l, wb2l);
  gnn_wb_pack<<<32, 256, 0, stream>>>(W2r, wb2r);

  const int tb = 256;
  gnn_degree<<<(E + tb - 1) / tb, tb, 0, stream>>>(dst, cnt, E);
  gnn_gcnt<<<(N + tb - 1) / tb, tb, 0, stream>>>(bat, gcnt, N);
  gnn_inv<<<(N + tb - 1) / tb, tb, 0, stream>>>(cnt, inv, N);

  const long long units   = (long long)E * 32;
  const int       sblocks = (int)((units + tb - 1) / tb);
  const int       mtiles  = (N + 15) / 16;
  const int       mblocks = (mtiles + 7) / 8;   // 8 row tiles per block

  // ---- layer 1: scatter + fused dual-GEMM + ReLU -> h1 ----
  gnn_scatter<<<sblocks, tb, 0, stream>>>(x, src, dst, agg, E);
  gnn_sage_gemm<<<mblocks, 256, 0, stream>>>(agg, inv, x, wb1l, wb1r, b1, h1,
                                             nullptr, nullptr, N, /*relu=*/1,
                                             /*pool=*/0);

  // ---- layer 2: re-zero agg, scatter h1, fused GEMM + pooled epilogue ----
  hipMemsetAsync(agg, 0, (size_t)N * GNN_D * 4, stream);
  gnn_scatter<<<sblocks, tb, 0, stream>>>(h1, src, dst, agg, E);
  gnn_sage_gemm<<<mblocks, 256, 0, stream>>>(agg, inv, h1, wb2l, wb2r, b2,
                                             nullptr, bat, gsum, N, /*relu=*/0,
                                             /*pool=*/1);

  // ---- global mean pool finalize ----
  gnn_finalize<<<(GNN_G * GNN_D + tb - 1) / tb, tb, 0, stream>>>(
      gsum, gcnt, (float*)d_out);
}
